// AttentionHead_2869038154395
// MI455X (gfx1250) — matrix operand order repaired and checked by reading, compiled, not run on hardware
//
#include <hip/hip_runtime.h>
#include <hip/hip_bf16.h>

typedef __attribute__((ext_vector_type(16))) _Float16 v16h;
typedef __attribute__((ext_vector_type(8)))  _Float16 v8h;
typedef __attribute__((ext_vector_type(4)))  _Float16 v4h;
typedef __attribute__((ext_vector_type(2)))  _Float16 h2;
typedef __attribute__((ext_vector_type(8)))  float    v8f;

// Problem constants (reference: B=16384, W=10, D=1024, K=V=64)
#define WSZ   10
#define DDIM  1024
#define NCOMB 192            // [k|q|v] combined output columns
#define BPWG  32             // batches per workgroup
#define ROWS  320            // BPWG * WSZ  (20 dense M-tiles; 2 per wave)
#define NTHR  320            // 10 waves
#define KC    128            // K-dim chunk staged in LDS
#define LWSH  136            // f16 row stride for weights [n][k]; 136*2 % 16 == 0
#define LXSH  136            // f16 row stride for x slab  [row][k]
#define KQSH  200            // phase-2 kqv f16 row stride (400 B % 16 == 0; 8-row de-alias)

#define LW_H   (NCOMB * LWSH)          // 26112 f16  (52,224 B) weights, K-transposed
#define LX_H   (ROWS * LXSH)           // 43520 f16  (87,040 B) x slab
#define PH1_BYTES (2 * (LW_H + LX_H))  // 139,264 B
#define KQV_BYTES (ROWS * KQSH * 2)    // 128,000 B (f16)
#define SC_BYTES  (ROWS * 12 * 4)      //  15,360 B
#define PH2_BYTES (KQV_BYTES + SC_BYTES)                                   // 143,360 B
#define SMEM_BYTES (PH2_BYTES > PH1_BYTES ? PH2_BYTES : PH1_BYTES)         // 143,360 B -> 2 WG/WGP

__device__ __forceinline__ v16h concat8(v8h lo, v8h hi) {
    return __builtin_shufflevector(lo, hi, 0,1,2,3,4,5,6,7,8,9,10,11,12,13,14,15);
}

__global__ __launch_bounds__(NTHR)
void attn_head_kernel(const float* __restrict__ x,
                      const float* __restrict__ wk,
                      const float* __restrict__ wq,
                      const float* __restrict__ wv,
                      float* __restrict__ out) {
    extern __shared__ char smem[];
    _Float16* lwh = (_Float16*)smem;            // [192][136] f16 weight chunk, [n][k]
    _Float16* lxh = (_Float16*)smem + LW_H;     // [320][136] f16 x chunk, [row][k]

    const int tid  = threadIdx.x;
    const int lane = tid & 31;
    const int wvid = tid >> 5;                  // wave id: owns M-tiles 2*wvid, 2*wvid+1
    const int lm   = lane & 15;
    const int hs   = lane >> 4;                 // half-wave select
    const long wgrow = (long)blockIdx.x * ROWS; // first flat (b*W+w) row of this WG

    v8f acc0[12] = {};                          // M-tile 0: 16x192 slab in f32
    v8f acc1[12] = {};                          // M-tile 1

    for (int kc = 0; kc < DDIM; kc += KC) {
        __syncthreads();
        // ---- stage weights: K-pair packed h2 stores matching B VGPR packing ----
        for (int idx = tid; idx < 3 * 64 * 16; idx += NTHR) {
            const int a   = idx >> 10;               // 0:wk 1:wq 2:wv
            const int rem = idx & 1023;
            const int rp  = rem >> 4;                // k-pair index (0..63)
            const int c   = (rem & 15) << 2;         // col quad base
            const float* wp = (a == 0) ? wk : (a == 1) ? wq : wv;
            const float4 fa = *(const float4*)(wp + (long)(kc + 2 * rp) * 64 + c);
            const float4 fb = *(const float4*)(wp + (long)(kc + 2 * rp + 1) * 64 + c);
            h2* dst = (h2*)(lwh) + rp;               // h2 stride per n-row = LWSH/2
            h2 p0; p0[0] = (_Float16)fa.x; p0[1] = (_Float16)fb.x; dst[(a*64 + c + 0) * (LWSH/2)] = p0;
            h2 p1; p1[0] = (_Float16)fa.y; p1[1] = (_Float16)fb.y; dst[(a*64 + c + 1) * (LWSH/2)] = p1;
            h2 p2; p2[0] = (_Float16)fa.z; p2[1] = (_Float16)fb.z; dst[(a*64 + c + 2) * (LWSH/2)] = p2;
            h2 p3; p3[0] = (_Float16)fa.w; p3[1] = (_Float16)fb.w; dst[(a*64 + c + 3) * (LWSH/2)] = p3;
        }
        // ---- stage x slab: 320 rows x 16 groups of 8 floats -> v8h b128 stores ----
        for (int idx = tid; idx < ROWS * (KC / 8); idx += NTHR) {
            const int r = idx >> 4;
            const int g = (idx & 15) << 3;
            const float* src = x + (wgrow + r) * DDIM + kc + g;
            const float4 f0 = *(const float4*)(src);
            const float4 f1 = *(const float4*)(src + 4);
            if (kc + KC < DDIM) __builtin_prefetch(src + KC, 0, 3);
            v8h h;
            h[0] = (_Float16)f0.x; h[1] = (_Float16)f0.y; h[2] = (_Float16)f0.z; h[3] = (_Float16)f0.w;
            h[4] = (_Float16)f1.x; h[5] = (_Float16)f1.y; h[6] = (_Float16)f1.z; h[7] = (_Float16)f1.w;
            *(v8h*)(lxh + r * LXSH + g) = h;
        }
        __syncthreads();

        // ---- WMMA: 2 M-tiles x 12 N-tiles; each B fragment feeds 2 WMMAs ----
        const _Float16* ar0 = lxh + (32 * wvid + lm) * LXSH + (hs << 4);
        const _Float16* ar1 = ar0 + 16 * LXSH;
        #pragma unroll
        for (int ks = 0; ks < KC; ks += 32) {
            const v16h A0 = concat8(*(const v8h*)(ar0 + ks), *(const v8h*)(ar0 + ks + 8));
            const v16h A1 = concat8(*(const v8h*)(ar1 + ks), *(const v8h*)(ar1 + ks + 8));
            const _Float16* bp = lwh + lm * LWSH + (hs << 4) + ks;
            #pragma unroll
            for (int g = 0; g < 3; ++g) {
                v8h bl[4], bh[4];
                #pragma unroll
                for (int u = 0; u < 4; ++u) {
                    const _Float16* np = bp + (g * 4 + u) * 16 * LWSH;
                    bl[u] = *(const v8h*)(np);
                    bh[u] = *(const v8h*)(np + 8);
                }
                #pragma unroll
                for (int u = 0; u < 4; ++u) {
                    const int nt = g * 4 + u;
                    const v16h Bv = concat8(bl[u], bh[u]);
                    acc0[nt] = __builtin_amdgcn_wmma_f32_16x16x32_f16(
                        false, A0, false, Bv, (short)0, acc0[nt], false, false);
                    acc1[nt] = __builtin_amdgcn_wmma_f32_16x16x32_f16(
                        false, A1, false, Bv, (short)0, acc1[nt], false, false);
                }
            }
        }
    }

    // ---------------- phase 2: attention epilogue (f16 kqv, fp32 math) ----------------
    _Float16* kqvh = (_Float16*)smem;                // [320][200] f16, overlays phase-1 LDS
    float* sc = (float*)(smem + KQV_BYTES);          // [320][12] scores -> softmax weights
    __syncthreads();
    #pragma unroll
    for (int nt = 0; nt < 12; ++nt) {
        #pragma unroll
        for (int r = 0; r < 8; ++r) {                // C/D layout: VGPR r -> row r + 8*hs
            const int row0 = 32 * wvid + (hs << 3) + r;
            const int col  = nt * 16 + lm;
            kqvh[row0 * KQSH + col] = (_Float16)acc0[nt][r];
            kqvh[(row0 + 16) * KQSH + col] = (_Float16)acc1[nt][r];
        }
    }
    __syncthreads();

    // scores[b][i][j] = k_i . q_j (fp32 acc), masked: j>i -> -1e5, else min(s, 1e5)
    for (int idx = tid; idx < BPWG * 100; idx += NTHR) {
        const int bb  = idx / 100;
        const int rem = idx - bb * 100;
        const int i   = rem / 10;
        const int j   = rem - i * 10;
        const _Float16* ki = kqvh + (bb * 10 + i) * KQSH;        // k cols 0..63
        const _Float16* qj = kqvh + (bb * 10 + j) * KQSH + 64;   // q cols 64..127
        float s = 0.f;
        #pragma unroll
        for (int c = 0; c < 64; c += 8) {
            const v8h kv = *(const v8h*)(ki + c);
            const v8h qv = *(const v8h*)(qj + c);
            #pragma unroll
            for (int e = 0; e < 8; ++e) s += (float)kv[e] * (float)qv[e];
        }
        s = (j > i) ? -1e5f : fminf(s, 1e5f);
        sc[(bb * 10 + i) * 12 + j] = s;
    }
    __syncthreads();

    // row softmax over j
    for (int r = tid; r < ROWS; r += NTHR) {
        float* srow = sc + r * 12;
        float m = srow[0];
        #pragma unroll
        for (int j = 1; j < 10; ++j) m = fmaxf(m, srow[j]);
        float e[10]; float sum = 0.f;
        #pragma unroll
        for (int j = 0; j < 10; ++j) { e[j] = expf(srow[j] - m); sum += e[j]; }
        const float inv = 1.f / sum;
        #pragma unroll
        for (int j = 0; j < 10; ++j) srow[j] = e[j] * inv;
    }
    __syncthreads();

    // out[row][vc..vc+3] = sum_j a[row][j] * v[j][vc..vc+3]  (v = cols 128..191)
    for (int idx = tid; idx < ROWS * 16; idx += NTHR) {
        const int r  = idx >> 4;
        const int vc = (idx & 15) << 2;
        const int bb = r / 10;
        const float* arow2 = sc + r * 12;
        const _Float16* vbase = kqvh + (bb * 10) * KQSH + 128 + vc;
        float4 o = make_float4(0.f, 0.f, 0.f, 0.f);
        #pragma unroll
        for (int j = 0; j < 10; ++j) {
            const float aj = arow2[j];
            const v4h vv = *(const v4h*)(vbase + j * KQSH);
            o.x += aj * (float)vv[0]; o.y += aj * (float)vv[1];
            o.z += aj * (float)vv[2]; o.w += aj * (float)vv[3];
        }
        *(float4*)(out + (wgrow + r) * 64 + vc) = o;
    }
}

extern "C" void kernel_launch(void* const* d_in, const int* in_sizes, int n_in,
                              void* d_out, int out_size, void* d_ws, size_t ws_size,
                              hipStream_t stream) {
    const float* x  = (const float*)d_in[0];
    const float* wk = (const float*)d_in[1];
    const float* wq = (const float*)d_in[2];
    const float* wv = (const float*)d_in[3];
    float* out = (float*)d_out;

    const int B = in_sizes[0] / (WSZ * DDIM);   // 16384
    const int grid = B / BPWG;                  // 512 workgroups, 10 waves each
    attn_head_kernel<<<grid, NTHR, SMEM_BYTES, stream>>>(x, wk, wq, wv, out);
}
